// IntelligentDownsampler_51470888075610
// MI455X (gfx1250) — matrix operand order repaired and checked by reading, not compiled, not run on hardware
//
#include <hip/hip_runtime.h>
#include <hip/hip_bf16.h>
#include <math.h>

// ---------------- problem constants (fixed by setup_inputs) ----------------
#define B_    2
#define N_    8192
#define C_    64
#define KC    64      // K_CAND
#define KF    16      // K_FINAL
#define NS    2048    // num_samples
#define NH    1024    // num_curv == num_fps
#define KPAD  96      // 67 padded to 3 k-tiles of 32
#define HID   128
#define EPSF  1e-8f

typedef __attribute__((ext_vector_type(16))) _Float16 v16h;
typedef __attribute__((ext_vector_type(8)))  _Float16 v8h;
typedef __attribute__((ext_vector_type(8)))  float    v8f;
typedef __attribute__((ext_vector_type(4)))  unsigned int u32x4;
typedef __attribute__((ext_vector_type(8)))  int      i32x8;
typedef __attribute__((ext_vector_type(4)))  int      i32x4;

#if defined(__gfx1250__) && __has_builtin(__builtin_amdgcn_tensor_load_to_lds) && \
    __has_builtin(__builtin_amdgcn_s_wait_tensorcnt)
#define USE_TDM 1
#endif

// ---------------------------------------------------------------------------
// Kernel 0: repack w1 (67x128 f32) into WMMA-B fragment-major f16 layout:
//   w1f[((ntile*3 + ktile)*32 + lane)*16 + e]
//   holds W1[k][n], k = ktile*32 + (lane&16)/2 + (e&7) + (e&8)*2, n = ntile*16 + (lane&15)
// so each lane's 32-byte B fragment is contiguous (2x global_load_b128).
// ---------------------------------------------------------------------------
__global__ void k_prep_w1(const float* __restrict__ w1, _Float16* __restrict__ w1f) {
    int i = blockIdx.x * blockDim.x + threadIdx.x;      // 0 .. 8*3*32*16-1
    if (i >= 8 * 3 * 32 * 16) return;
    int e    = i & 15;
    int lane = (i >> 4) & 31;
    int t    = i >> 9;            // 0..23
    int kt   = t % 3;
    int w    = t / 3;
    int k = kt * 32 + ((lane & 16) ? 8 : 0) + (e & 7) + ((e & 8) ? 16 : 0);
    int n = w * 16 + (lane & 15);
    w1f[i] = (k < 67) ? (_Float16)w1[k * HID + n] : (_Float16)0.0f;
}

// ---------------------------------------------------------------------------
// Kernel 1: global max over xyz -> far_val = max + 1
// ---------------------------------------------------------------------------
__global__ void k_maxxyz(const float* __restrict__ xyz, float* __restrict__ farval) {
    __shared__ float red[256];
    int tid = threadIdx.x;
    float m = -3.0e38f;
    for (int i = tid; i < B_ * N_ * 3; i += 256) m = fmaxf(m, xyz[i]);
    red[tid] = m; __syncthreads();
    for (int s = 128; s > 0; s >>= 1) {
        if (tid < s) red[tid] = fmaxf(red[tid], red[tid + s]);
        __syncthreads();
    }
    if (tid == 0) farval[0] = red[0] + 1.0f;
}

// ---------------------------------------------------------------------------
// Kernel 2: KNN — per-thread query point, LDS-tiled candidates, top-64 list.
// ---------------------------------------------------------------------------
__global__ void k_knn(const float* __restrict__ xyz, int* __restrict__ cand) {
    __shared__ float tp[256 * 3];
    int tid = threadIdx.x;
    int q   = blockIdx.x * 256 + tid;
    int b   = blockIdx.y;
    const float* bx = xyz + (size_t)b * N_ * 3;

    float qx = bx[q * 3 + 0], qy = bx[q * 3 + 1], qz = bx[q * 3 + 2];

    float hd[KC]; int hi[KC];
    #pragma unroll
    for (int t = 0; t < KC; ++t) { hd[t] = 3.0e38f; hi[t] = 0; }
    float worstV = 3.0e38f; int worstP = 0;

    for (int base = 0; base < N_; base += 256) {
        __syncthreads();
        tp[tid * 3 + 0] = bx[(base + tid) * 3 + 0];
        tp[tid * 3 + 1] = bx[(base + tid) * 3 + 1];
        tp[tid * 3 + 2] = bx[(base + tid) * 3 + 2];
        __syncthreads();
        for (int jj = 0; jj < 256; ++jj) {
            int j = base + jj;
            if (j == q) continue;
            float dx = tp[jj * 3 + 0] - qx;
            float dy = tp[jj * 3 + 1] - qy;
            float dz = tp[jj * 3 + 2] - qz;
            float d2 = dx * dx + dy * dy + dz * dz;
            if (d2 < worstV) {
                hd[worstP] = d2; hi[worstP] = j;
                float wv = -1.0f; int wp = 0;
                #pragma unroll 8
                for (int t = 0; t < KC; ++t)
                    if (hd[t] > wv) { wv = hd[t]; wp = t; }
                worstV = wv; worstP = wp;
            }
        }
    }
    int* out = cand + ((size_t)b * N_ + q) * KC;
    #pragma unroll 8
    for (int t = 0; t < KC; ++t) out[t] = hi[t];
}

// A 16x32 fragment from LDS: two aligned 16B runs -> 2x ds_load_b128
__device__ __forceinline__ v16h load_a_frag(const _Float16* Asm, int lane,
                                            int mtile, int kbase) {
    int M  = mtile * 16 + (lane & 15);
    int ko = kbase + ((lane & 16) ? 8 : 0);
    const _Float16* p = Asm + M * KPAD + ko;
    v8h lo = *(const v8h*)p;
    v8h hi = *(const v8h*)(p + 16);
    return __builtin_shufflevector(lo, hi, 0, 1, 2, 3, 4, 5, 6, 7,
                                           8, 9, 10, 11, 12, 13, 14, 15);
}

__device__ __forceinline__ float gelu_exact(float x) {
    return 0.5f * x * (1.0f + erff(x * 0.70710678118654752f));
}

// ---------------------------------------------------------------------------
// Kernel 3: per-point MLP scoring (WMMA), top-16, 3x3 cov eigen -> curvature,
// feature-variance. One workgroup (8 waves) per point.
// Neighbor feature rows gathered by the Tensor Data Mover (gather-mode D#,
// 16 rows x 16-bit indices per descriptor, 4 descriptors), wave 0 issues.
// ---------------------------------------------------------------------------
__global__ void __launch_bounds__(256)
k_mlp(const float* __restrict__ xyz, const float* __restrict__ feat,
      const _Float16* __restrict__ w1f, const float* __restrict__ b1,
      const float* __restrict__ w2, const float* __restrict__ b2,
      const int* __restrict__ cand,
      float* __restrict__ curvb, float* __restrict__ fvarb) {
    __shared__ _Float16 Asm[KC * KPAD];      // 12 KB  (A: 64 x 96 f16)
    __shared__ float featbuf[KC * C_];       // 16 KB  (gathered neighbor rows)
    __shared__ float qfs[C_];
    __shared__ float scores[KC];
    __shared__ float relx[KC * 3];
    __shared__ float nrm2s[KC];
    __shared__ float w2s[HID];
    __shared__ int   sel[KF];
    __shared__ float covm[6];
    __shared__ float fvacc;

    const int tid  = threadIdx.x;
    const int lane = tid & 31;
    const int wave = tid >> 5;               // 0..7 -> output N-tile
    const int n    = blockIdx.x;
    const int b    = blockIdx.y;

    __builtin_prefetch(w1f, 0, 1);           // global_prefetch_b8 (L2-resident)

    const int*   cd = cand + ((size_t)b * N_ + n) * KC;
    const float* qx = xyz  + ((size_t)b * N_ + n) * 3;
    const float* qf = feat + ((size_t)b * N_ + n) * C_;

    if (tid < KC)  scores[tid] = b2[0];
    if (tid < C_)  qfs[tid]    = qf[tid];
    if (tid < HID) w2s[tid]    = w2[tid];

    // ---- gather 64 neighbor feature rows (256B each) into LDS via TDM ----
#ifdef USE_TDM
    if (wave == 0) {
        int c0 = cd[2 * lane + 0];
        int c1 = cd[2 * lane + 1];
        int packed = (c0 & 0xFFFF) | (c1 << 16);      // lane l: rows 2l, 2l+1
        unsigned ldsbase = (unsigned)(uintptr_t)(void*)featbuf;   // LDS = addr[31:0]
        unsigned long long gbase =
            (unsigned long long)(uintptr_t)(feat + (size_t)b * N_ * C_);
        i32x8 g1 = {};
        g1[0] = 0x00020000;                    // data_size = 2 (4 bytes/elem)
        g1[1] = (int)(((unsigned)C_) << 16);   // tensor_dim0 = 64
        g1[2] = (int)(((unsigned)N_) << 16);   // tensor_dim1 = 8192
        g1[3] = (int)(((unsigned)C_) << 16);   // tile_dim0   = 64
        g1[4] = 16;                            // tile_dim1   = 16 valid indices
        g1[5] = C_;                            // tensor_dim0_stride = 64
        g1[6] = 0; g1[7] = 0;
        i32x8 gz = {};                         // unused trailing descriptor group
        #pragma unroll
        for (int g = 0; g < 4; ++g) {
            u32x4 g0 = {};
            g0[0] = 0x80000001u;               // gather_mode | 16-bit idx | count=1
            g0[1] = ldsbase + (unsigned)g * (16u * C_ * 4u);
            g0[2] = (unsigned)gbase;
            g0[3] = ((unsigned)(gbase >> 32) & 0x01FFFFFFu) | 0x80000000u; // type=2
            i32x4 gi0, gi1;
            gi0[0] = __builtin_amdgcn_readlane(packed, 8 * g + 0);
            gi0[1] = __builtin_amdgcn_readlane(packed, 8 * g + 1);
            gi0[2] = __builtin_amdgcn_readlane(packed, 8 * g + 2);
            gi0[3] = __builtin_amdgcn_readlane(packed, 8 * g + 3);
            gi1[0] = __builtin_amdgcn_readlane(packed, 8 * g + 4);
            gi1[1] = __builtin_amdgcn_readlane(packed, 8 * g + 5);
            gi1[2] = __builtin_amdgcn_readlane(packed, 8 * g + 6);
            gi1[3] = __builtin_amdgcn_readlane(packed, 8 * g + 7);
            __builtin_amdgcn_tensor_load_to_lds(g0, g1, gi0, gi1, gz, 0);
        }
        __builtin_amdgcn_s_wait_tensorcnt(0);
    }
#else
    for (int i = tid; i < KC * C_; i += 256) {
        int k = i >> 6, c = i & 63;
        featbuf[i] = feat[((size_t)b * N_ + cd[k]) * C_ + c];
    }
#endif
    __syncthreads();

    // ---- build A (rel_xyz ++ rel_feat, zero pad), relx cache, ||rel_feat||^2
    {
        int k   = tid >> 2;                  // candidate row 0..63
        int sub = tid & 3;
        int idx = cd[k];
        const float* nx = xyz + ((size_t)b * N_ + idx) * 3;
        const float* fb = featbuf + k * C_;
        float ss = 0.0f;
        for (int c = sub; c < KPAD; c += 4) {
            float v;
            if (c < 3)       { v = nx[c] - qx[c]; relx[k * 3 + c] = v; }
            else if (c < 67) { v = fb[c - 3] - qfs[c - 3]; ss += v * v; }
            else               v = 0.0f;
            Asm[k * KPAD + c] = (_Float16)v;
        }
        ss += __shfl_xor(ss, 1, 4);
        ss += __shfl_xor(ss, 2, 4);
        if (sub == 0) nrm2s[k] = ss;
    }
    __syncthreads();

    // ---- GEMM: h = gelu(A @ W1 + b1); score = h @ w2 + b2 ----
    {
        const v16h* wf = (const v16h*)w1f;   // fragment-major: 2x global_load_b128
        v16h bfr0 = wf[(wave * 3 + 0) * 32 + lane];
        v16h bfr1 = wf[(wave * 3 + 1) * 32 + lane];
        v16h bfr2 = wf[(wave * 3 + 2) * 32 + lane];
        const int ncol = wave * 16 + (lane & 15);
        const float w2v = w2s[ncol];
        const float bia = b1[ncol];
        #pragma unroll
        for (int mt = 0; mt < 4; ++mt) {
            v8f c = {};
            v16h a0 = load_a_frag(Asm, lane, mt, 0);
            c = __builtin_amdgcn_wmma_f32_16x16x32_f16(false, a0, false, bfr0,
                                                       (short)0, c, false, false);
            v16h a1 = load_a_frag(Asm, lane, mt, 32);
            c = __builtin_amdgcn_wmma_f32_16x16x32_f16(false, a1, false, bfr1,
                                                       (short)0, c, false, false);
            v16h a2 = load_a_frag(Asm, lane, mt, 64);
            c = __builtin_amdgcn_wmma_f32_16x16x32_f16(false, a2, false, bfr2,
                                                       (short)0, c, false, false);
            float part[8];
            #pragma unroll
            for (int r = 0; r < 8; ++r)
                part[r] = gelu_exact(c[r] + bia) * w2v;
            #pragma unroll
            for (int m = 1; m < 16; m <<= 1)
                #pragma unroll
                for (int r = 0; r < 8; ++r)
                    part[r] += __shfl_xor(part[r], m, 16);
            if ((lane & 15) == 0) {
                int rbase = mt * 16 + ((lane & 16) ? 8 : 0);
                #pragma unroll
                for (int r = 0; r < 8; ++r)
                    atomicAdd(&scores[rbase + r], part[r]);   // ds_add_f32
            }
        }
    }
    __syncthreads();

    // ---- top-16 of 64 scores (wave 0, butterfly argmax, stable ties) ----
    if (wave == 0) {
        float s0 = scores[lane], s1 = scores[lane + 32];
        bool a0 = true, a1 = true;
        for (int it = 0; it < KF; ++it) {
            float v = -3.0e38f; int id = 1 << 20;
            if (a0) { v = s0; id = lane; }
            if (a1 && (s1 > v || (s1 == v && lane + 32 < id))) { v = s1; id = lane + 32; }
            #pragma unroll
            for (int m = 1; m < 32; m <<= 1) {
                float ov = __shfl_xor(v, m, 32);
                int  oid = __shfl_xor(id, m, 32);
                if (ov > v || (ov == v && oid < id)) { v = ov; id = oid; }
            }
            if (id == lane)           a0 = false;
            else if (id == lane + 32) a1 = false;
            if (lane == 0) sel[it] = id;
        }
    }
    if (tid < 6)   covm[tid] = 0.0f;
    if (tid == 6)  fvacc = 0.0f;
    __syncthreads();

    if (tid < KF) {
        int k = sel[tid];
        float x = relx[k * 3 + 0], y = relx[k * 3 + 1], z = relx[k * 3 + 2];
        atomicAdd(&covm[0], x * x); atomicAdd(&covm[1], x * y);
        atomicAdd(&covm[2], x * z); atomicAdd(&covm[3], y * y);
        atomicAdd(&covm[4], y * z); atomicAdd(&covm[5], z * z);
        atomicAdd(&fvacc, sqrtf(nrm2s[k]));
    }
    __syncthreads();

    if (tid == 0) {
        const float inv = 1.0f / (float)KF;
        float a00 = covm[0] * inv, a01 = covm[1] * inv, a02 = covm[2] * inv;
        float a11 = covm[3] * inv, a12 = covm[4] * inv, a22 = covm[5] * inv;
        float p1 = a01 * a01 + a02 * a02 + a12 * a12;
        float q  = (a00 + a11 + a22) / 3.0f;
        float p2 = (a00 - q) * (a00 - q) + (a11 - q) * (a11 - q)
                 + (a22 - q) * (a22 - q) + 2.0f * p1;
        float e0, e1, e2;
        if (p2 < 1e-30f) { e0 = e1 = e2 = q; }
        else {
            float p  = sqrtf(p2 / 6.0f);
            float ip = 1.0f / p;
            float b00 = (a00 - q) * ip, b11 = (a11 - q) * ip, b22 = (a22 - q) * ip;
            float b01 = a01 * ip, b02 = a02 * ip, b12 = a12 * ip;
            float detB = b00 * (b11 * b22 - b12 * b12)
                       - b01 * (b01 * b22 - b12 * b02)
                       + b02 * (b01 * b12 - b11 * b02);
            float r = fminf(fmaxf(detB * 0.5f, -1.0f), 1.0f);
            float phi = acosf(r) / 3.0f;
            e0 = q + 2.0f * p * cosf(phi);
            e2 = q + 2.0f * p * cosf(phi + 2.0943951023931953f);
            e1 = 3.0f * q - e0 - e2;
        }
        float l2 = e0 * e0, l1 = e1 * e1, l0 = e2 * e2;
        curvb[(size_t)b * N_ + n] = l0 / (l0 + l1 + l2 + EPSF);
        fvarb[(size_t)b * N_ + n] = fvacc * inv;
    }
}

// ---------------------------------------------------------------------------
// Kernel 4: znorm(curv) + 0.5*znorm(feat_var) -> importance  (one WG / batch)
// ---------------------------------------------------------------------------
__global__ void k_importance(const float* __restrict__ curvb,
                             const float* __restrict__ fvarb,
                             float* __restrict__ imp) {
    __shared__ float sc[256], sc2[256], sf[256], sf2[256];
    int tid = threadIdx.x, b = blockIdx.x;
    const float* cb = curvb + (size_t)b * N_;
    const float* fb = fvarb + (size_t)b * N_;
    float c1 = 0, c2 = 0, f1 = 0, f2 = 0;
    for (int i = tid; i < N_; i += 256) {
        float c = cb[i], f = fb[i];
        c1 += c; c2 += c * c; f1 += f; f2 += f * f;
    }
    sc[tid] = c1; sc2[tid] = c2; sf[tid] = f1; sf2[tid] = f2;
    __syncthreads();
    for (int s = 128; s > 0; s >>= 1) {
        if (tid < s) {
            sc[tid] += sc[tid + s]; sc2[tid] += sc2[tid + s];
            sf[tid] += sf[tid + s]; sf2[tid] += sf2[tid + s];
        }
        __syncthreads();
    }
    float mc = sc[0] / N_;
    float vc = fmaxf((sc2[0] - (float)N_ * mc * mc) / (float)(N_ - 1), 0.0f);
    float dc = sqrtf(vc) + EPSF;
    float mf = sf[0] / N_;
    float vf = fmaxf((sf2[0] - (float)N_ * mf * mf) / (float)(N_ - 1), 0.0f);
    float df = sqrtf(vf) + EPSF;
    float* ob = imp + (size_t)b * N_;
    for (int i = tid; i < N_; i += 256)
        ob[i] = (cb[i] - mc) / dc + 0.5f * (fb[i] - mf) / df;
}

// ---------------------------------------------------------------------------
// Kernel 5: top-1024 of importance (destructive), build masked_xyz + mindist
// ---------------------------------------------------------------------------
__global__ void k_topk_mask(float* __restrict__ imp, const float* __restrict__ xyz,
                            const float* __restrict__ farval,
                            int* __restrict__ cidx, float* __restrict__ maskx,
                            float* __restrict__ mind) {
    __shared__ float sv[256];
    __shared__ int   si[256];
    int tid = threadIdx.x, b = blockIdx.x;
    float* ib = imp + (size_t)b * N_;
    const float* xb = xyz + (size_t)b * N_ * 3;
    float* mb = maskx + (size_t)b * N_ * 3;
    float* db = mind + (size_t)b * N_;
    for (int i = tid; i < N_; i += 256) {
        mb[i * 3 + 0] = xb[i * 3 + 0];
        mb[i * 3 + 1] = xb[i * 3 + 1];
        mb[i * 3 + 2] = xb[i * 3 + 2];
        db[i] = 1.0e10f;
    }
    __syncthreads();
    for (int it = 0; it < NH; ++it) {
        float bv = -3.0e38f; int bi = 1 << 20;
        for (int j = tid; j < N_; j += 256) {
            float v = ib[j];
            if (v > bv) { bv = v; bi = j; }
        }
        sv[tid] = bv; si[tid] = bi;
        __syncthreads();
        for (int s = 128; s > 0; s >>= 1) {
            if (tid < s) {
                if (sv[tid + s] > sv[tid] ||
                    (sv[tid + s] == sv[tid] && si[tid + s] < si[tid])) {
                    sv[tid] = sv[tid + s]; si[tid] = si[tid + s];
                }
            }
            __syncthreads();
        }
        if (tid == 0) { cidx[b * NH + it] = si[0]; ib[si[0]] = -3.0e38f; }
        __syncthreads();
    }
    float fv = farval[0];
    for (int i = tid; i < NH; i += 256) {
        int id = cidx[b * NH + i];
        mb[id * 3 + 0] = fv; mb[id * 3 + 1] = fv; mb[id * 3 + 2] = fv;
    }
}

// ---------------------------------------------------------------------------
// Kernel 6: farthest point sampling (sequential 1024 iters, one WG / batch)
// ---------------------------------------------------------------------------
__global__ void k_fps(const float* __restrict__ maskx, float* __restrict__ mind,
                      int* __restrict__ fidx) {
    __shared__ float sv[256];
    __shared__ int   si[256];
    __shared__ int   slast;
    __shared__ float lp[3];
    int tid = threadIdx.x, b = blockIdx.x;
    const float* mb = maskx + (size_t)b * N_ * 3;
    float* db = mind + (size_t)b * N_;
    if (tid == 0) slast = 0;
    __syncthreads();
    for (int it = 0; it < NH; ++it) {
        if (tid == 0) {
            int l = slast;
            fidx[b * NH + it] = l;
            lp[0] = mb[l * 3 + 0]; lp[1] = mb[l * 3 + 1]; lp[2] = mb[l * 3 + 2];
        }
        __syncthreads();
        float lx = lp[0], ly = lp[1], lz = lp[2];
        float bv = -3.0e38f; int bi = 1 << 20;
        for (int j = tid; j < N_; j += 256) {
            float dx = mb[j * 3 + 0] - lx;
            float dy = mb[j * 3 + 1] - ly;
            float dz = mb[j * 3 + 2] - lz;
            float d  = dx * dx + dy * dy + dz * dz;
            float md = fminf(db[j], d);
            db[j] = md;
            if (md > bv) { bv = md; bi = j; }
        }
        sv[tid] = bv; si[tid] = bi;
        __syncthreads();
        for (int s = 128; s > 0; s >>= 1) {
            if (tid < s) {
                if (sv[tid + s] > sv[tid] ||
                    (sv[tid + s] == sv[tid] && si[tid + s] < si[tid])) {
                    sv[tid] = sv[tid + s]; si[tid] = si[tid + s];
                }
            }
            __syncthreads();
        }
        if (tid == 0) slast = si[0];
        __syncthreads();
    }
}

// ---------------------------------------------------------------------------
// Kernel 7: gather outputs: sampled_xyz (B*2048*3 f32) ++ merged (B*2048 i32)
// ---------------------------------------------------------------------------
__global__ void k_gather(const float* __restrict__ xyz, const int* __restrict__ cidx,
                         const int* __restrict__ fidx, float* __restrict__ outx,
                         int* __restrict__ outi) {
    int j = blockIdx.x * blockDim.x + threadIdx.x;
    if (j >= B_ * NS) return;
    int b = j / NS, r = j % NS;
    int m = (r < NH) ? cidx[b * NH + r] : fidx[b * NH + (r - NH)];
    const float* p = xyz + ((size_t)b * N_ + m) * 3;
    outx[j * 3 + 0] = p[0];
    outx[j * 3 + 1] = p[1];
    outx[j * 3 + 2] = p[2];
    outi[j] = m;
}

// ---------------------------------------------------------------------------
extern "C" void kernel_launch(void* const* d_in, const int* in_sizes, int n_in,
                              void* d_out, int out_size, void* d_ws, size_t ws_size,
                              hipStream_t stream) {
    const float* xyz  = (const float*)d_in[0];
    const float* feat = (const float*)d_in[1];
    const float* w1   = (const float*)d_in[2];
    const float* b1   = (const float*)d_in[3];
    const float* w2   = (const float*)d_in[4];
    const float* b2   = (const float*)d_in[5];
    (void)in_sizes; (void)n_in; (void)out_size; (void)ws_size;

    char* ws = (char*)d_ws;
    size_t off = 0;
    auto carve = [&](size_t bytes) { char* p = ws + off; off = (off + bytes + 255) & ~(size_t)255; return p; };
    int*      cand  = (int*)      carve((size_t)B_ * N_ * KC * 4);
    _Float16* w1f   = (_Float16*) carve((size_t)8 * 3 * 32 * 16 * 2);
    float*    curvb = (float*)    carve((size_t)B_ * N_ * 4);
    float*    fvarb = (float*)    carve((size_t)B_ * N_ * 4);
    float*    imp   = (float*)    carve((size_t)B_ * N_ * 4);
    float*    maskx = (float*)    carve((size_t)B_ * N_ * 3 * 4);
    float*    mind  = (float*)    carve((size_t)B_ * N_ * 4);
    int*      cidx  = (int*)      carve((size_t)B_ * NH * 4);
    int*      fidx  = (int*)      carve((size_t)B_ * NH * 4);
    float*    farv  = (float*)    carve(256);

    float* outx = (float*)d_out;
    int*   outi = (int*)((float*)d_out + (size_t)B_ * NS * 3);

    k_prep_w1   <<<(8 * 3 * 32 * 16 + 255) / 256, 256, 0, stream>>>(w1, w1f);
    k_maxxyz    <<<1, 256, 0, stream>>>(xyz, farv);
    k_knn       <<<dim3(N_ / 256, B_), 256, 0, stream>>>(xyz, cand);
    k_mlp       <<<dim3(N_, B_), 256, 0, stream>>>(xyz, feat, w1f, b1, w2, b2,
                                                   cand, curvb, fvarb);
    k_importance<<<B_, 256, 0, stream>>>(curvb, fvarb, imp);
    k_topk_mask <<<B_, 256, 0, stream>>>(imp, xyz, farv, cidx, maskx, mind);
    k_fps       <<<B_, 256, 0, stream>>>(maskx, mind, fidx);
    k_gather    <<<(B_ * NS + 255) / 256, 256, 0, stream>>>(xyz, cidx, fidx, outx, outi);
}
